// GraphTransformer_53764400611517
// MI455X (gfx1250) — matrix operand order repaired and checked by reading, not compiled, not run on hardware
//
#include <hip/hip_runtime.h>
#include <hip/hip_bf16.h>

// ---------- CDNA5 WMMA types ----------
typedef __attribute__((ext_vector_type(16))) __bf16   v16bf;
typedef __attribute__((ext_vector_type(8)))  float    v8f;
typedef __attribute__((ext_vector_type(4)))  unsigned uint4v;
typedef __attribute__((ext_vector_type(8)))  unsigned uint8v;
typedef __attribute__((ext_vector_type(4)))  float    f4;

__device__ __forceinline__ unsigned short f2bf(float f) {
    unsigned u = __float_as_uint(f);
    unsigned r = u + 0x7FFFu + ((u >> 16) & 1u);   // round-to-nearest-even
    return (unsigned short)(r >> 16);
}

// ---------- embedding gather: h_bf16[N,256] = bf16(emb[x[n]]) ----------
__global__ void embed_kernel(const int* __restrict__ x, const float* __restrict__ emb,
                             unsigned short* __restrict__ hbf, int N) {
    long i = (long)blockIdx.x * blockDim.x + threadIdx.x;
    if (i >= (long)N * 256) return;
    int n = (int)(i >> 8), c = (int)(i & 255);
    hbf[i] = f2bf(emb[(size_t)x[n] * 256 + c]);
}

// ---------- pack fp32 W[K,N] into bf16 WMMA B-fragments ----------
// Layout (matches 16-bit A table order): fragment for tile
// (kt,nt): lane L holds column n = nt*16 + (L&15), element e holds
// k = kt*32 + (L>=16 ? 8 : 0) + (e&7) + (e>=8 ? 16 : 0).
__global__ void pack_w_kernel(const float* __restrict__ W, unsigned short* __restrict__ Wp,
                              int K, int N) {
    int idx = blockIdx.x * blockDim.x + threadIdx.x;
    if (idx >= K * N) return;
    int e    = idx & 15;
    int lane = (idx >> 4) & 31;
    int frag = idx >> 9;
    int ntiles = N >> 4;
    int nt = frag % ntiles, kt = frag / ntiles;
    int n = nt * 16 + (lane & 15);
    int k = kt * 32 + ((lane >= 16) ? 8 : 0) + (e & 7) + ((e & 8) ? 16 : 0);
    Wp[idx] = f2bf(W[(size_t)k * N + n]);
}

// ---------- WMMA GEMM: C[M,N] = bf16A[M,K] @ Wp + bias ----------
// A row-major bf16.  A-fragment per ISA 16-bit A 16x32 layout:
// lane L: m = L&15, elements 0..7 -> k = kb + (L>=16?8:0) + 0..7 (contiguous),
// elements 8..15 -> k = kb + 16 + (L>=16?8:0) + 0..7 (contiguous) => 2x b128.
__global__ void gemm_wmma_kernel(const unsigned short* __restrict__ A,
                                 const unsigned short* __restrict__ Wp,
                                 const float* __restrict__ bias,
                                 float* __restrict__ C,
                                 int M, int N, int K) {
    const int lane  = threadIdx.x & 31;
    const int wave  = threadIdx.x >> 5;
    const int mtile = blockIdx.x;
    const int ntile = blockIdx.y * 8 + wave;
    if (ntile * 16 >= N) return;                       // uniform per wave
    const int m    = mtile * 16 + (lane & 15);
    const int koff = (lane >> 4) * 8;
    const int ntiles = N >> 4;
    const unsigned short* arow = A + (size_t)m * K;

    v8f acc = {};
    for (int kb = 0; kb < K; kb += 32) {
        uint8v au, bu;
        au.lo = *(const uint4v*)(arow + kb + koff);
        au.hi = *(const uint4v*)(arow + kb + 16 + koff);
        const unsigned short* bp =
            Wp + ((size_t)((kb >> 5) * ntiles + ntile) * 32 + lane) * 16;
        bu.lo = *(const uint4v*)(bp);
        bu.hi = *(const uint4v*)(bp + 8);
#if defined(__HIP_DEVICE_COMPILE__)
        __builtin_prefetch(arow + kb + 64, 0, 3);      // global_prefetch
#endif
        v16bf av = __builtin_bit_cast(v16bf, au);
        v16bf bv = __builtin_bit_cast(v16bf, bu);
        acc = __builtin_amdgcn_wmma_f32_16x16x32_bf16(
            /*neg_a=*/false, av, /*neg_b=*/false, bv,
            /*c_mod=*/(short)0, acc, /*reuse_a=*/false, /*reuse_b=*/false);
    }
    const int n  = ntile * 16 + (lane & 15);
    const float bn = bias ? bias[n] : 0.f;
    const int mbase = mtile * 16 + (lane >> 4) * 8;    // D: lanes 16-31 hold M=r+8
#pragma unroll
    for (int r = 0; r < 8; ++r)
        C[(size_t)(mbase + r) * N + n] = acc[r] + bn;
}

// ---------- per-edge attention logits + segment max (wave per (e,h)) ----------
__global__ void edge_logits_kernel(const float* __restrict__ q, const float* __restrict__ k,
                                   const int* __restrict__ src, const int* __restrict__ dst,
                                   float* __restrict__ logits, unsigned* __restrict__ mkey, int E) {
    int e = blockIdx.x, h = threadIdx.x >> 5, lane = threadIdx.x & 31;
    if (e >= E) return;
    int s = src[e], d = dst[e];
    f4 qv = *(const f4*)(q + (size_t)d * 1024 + h * 128 + lane * 4);
    f4 kv = *(const f4*)(k + (size_t)s * 1024 + h * 128 + lane * 4);
    float acc = qv[0]*kv[0] + qv[1]*kv[1] + qv[2]*kv[2] + qv[3]*kv[3];
    for (int off = 16; off > 0; off >>= 1) acc += __shfl_xor(acc, off, 32);
    acc *= 0.08838834764831845f;                       // 1/sqrt(128)
    if (lane == 0) {
        logits[(size_t)e * 8 + h] = acc;
        unsigned b = __float_as_uint(acc);
        unsigned key = (b & 0x80000000u) ? ~b : (b | 0x80000000u); // order-preserving
        atomicMax(mkey + (size_t)d * 8 + h, key);
    }
}

__global__ void decode_m_kernel(const unsigned* __restrict__ mkey, float* __restrict__ m, int n) {
    int i = blockIdx.x * blockDim.x + threadIdx.x;
    if (i >= n) return;
    unsigned key = mkey[i];
    unsigned b = (key & 0x80000000u) ? (key & 0x7fffffffu) : ~key;
    bool fin = ((b & 0x7f800000u) != 0x7f800000u);
    m[i] = fin ? __uint_as_float(b) : 0.f;             // matches jnp.where(isfinite)
}

__global__ void ex_den_kernel(float* __restrict__ logits, const float* __restrict__ m,
                              const int* __restrict__ dst, float* __restrict__ den, int EH) {
    int i = blockIdx.x * blockDim.x + threadIdx.x;
    if (i >= EH) return;
    int e = i >> 3, h = i & 7;
    int d = dst[e];
    float ex = __expf(logits[i] - m[(size_t)d * 8 + h]);
    logits[i] = ex;                                    // reuse as ex
    atomicAdd(den + (size_t)d * 8 + h, ex);
}

// ---------- scatter alpha * v into agg[dst] (wave per (e,h)) ----------
__global__ void scatter_kernel(const float* __restrict__ ex, const float* __restrict__ den,
                               const float* __restrict__ v, const int* __restrict__ src,
                               const int* __restrict__ dst, float* __restrict__ agg, int E) {
    int e = blockIdx.x, h = threadIdx.x >> 5, lane = threadIdx.x & 31;
    if (e >= E) return;
    int s = src[e], d = dst[e];
    float alpha = ex[(size_t)e * 8 + h] / (den[(size_t)d * 8 + h] + 1e-16f);
    f4 vv = *(const f4*)(v + (size_t)s * 1024 + h * 128 + lane * 4);
    float* ap = agg + (size_t)d * 1024 + h * 128 + lane * 4;
    atomicAdd(ap + 0, alpha * vv[0]);
    atomicAdd(ap + 1, alpha * vv[1]);
    atomicAdd(ap + 2, alpha * vv[2]);
    atomicAdd(ap + 3, alpha * vv[3]);
}

// ---------- fused: head-mean, beta gate, (+res), layernorm, relu ----------
__global__ void node_finalize_kernel(const float* __restrict__ agg, const float* __restrict__ xr,
                                     const float* __restrict__ Wb, const float* __restrict__ g,
                                     const float* __restrict__ bLN, const float* res,
                                     float* hout, unsigned short* __restrict__ hbf, int N) {
    int node = (int)(((long)blockIdx.x * blockDim.x + threadIdx.x) >> 5);
    int lane = threadIdx.x & 31;
    if (node >= N) return;
    int c0 = lane * 4;
    float o[4], x_r[4];
#pragma unroll
    for (int j = 0; j < 4; ++j) {
        float s = 0.f;
#pragma unroll
        for (int h = 0; h < 8; ++h) s += agg[(size_t)node * 1024 + h * 128 + c0 + j];
        o[j]  = s * 0.125f;                            // head mean
        x_r[j] = xr[(size_t)node * 128 + c0 + j];
    }
    float part = 0.f;
#pragma unroll
    for (int j = 0; j < 4; ++j)
        part += o[j] * Wb[c0 + j] + x_r[j] * Wb[128 + c0 + j] + (o[j] - x_r[j]) * Wb[256 + c0 + j];
    for (int off = 16; off > 0; off >>= 1) part += __shfl_xor(part, off, 32);
    float beta = 1.f / (1.f + __expf(-part));
    float hv[4];
#pragma unroll
    for (int j = 0; j < 4; ++j) {
        hv[j] = beta * x_r[j] + (1.f - beta) * o[j];
        if (res) hv[j] += res[(size_t)node * 128 + c0 + j];
    }
    float s1 = hv[0] + hv[1] + hv[2] + hv[3];
    for (int off = 16; off > 0; off >>= 1) s1 += __shfl_xor(s1, off, 32);
    float mu = s1 * (1.f / 128.f);
    float s2 = 0.f;
#pragma unroll
    for (int j = 0; j < 4; ++j) { float dd = hv[j] - mu; s2 += dd * dd; }
    for (int off = 16; off > 0; off >>= 1) s2 += __shfl_xor(s2, off, 32);
    float rinv = rsqrtf(s2 * (1.f / 128.f) + 1e-5f);
#pragma unroll
    for (int j = 0; j < 4; ++j) {
        float y = (hv[j] - mu) * rinv * g[c0 + j] + bLN[c0 + j];
        y = fmaxf(y, 0.f);
        hout[(size_t)node * 128 + c0 + j] = y;
        hbf[(size_t)node * 128 + c0 + j] = f2bf(y);
    }
}

// ---------- global mean pool (atomics) ----------
__global__ void pool_kernel(const float* __restrict__ h, const int* __restrict__ batch,
                            float* __restrict__ pooled, float* __restrict__ cnt, int N) {
    long i = (long)blockIdx.x * blockDim.x + threadIdx.x;
    int node = (int)(i >> 5), lane = (int)(i & 31);
    if (node >= N) return;
    int gix = batch[node];
    f4 hv = *(const f4*)(h + (size_t)node * 128 + lane * 4);
    float* p = pooled + (size_t)gix * 128 + lane * 4;
    atomicAdd(p + 0, hv[0]); atomicAdd(p + 1, hv[1]);
    atomicAdd(p + 2, hv[2]); atomicAdd(p + 3, hv[3]);
    if (lane == 0) atomicAdd(cnt + gix, 1.f);
}

// ---------- classifier MLP: 64 graphs, 128->64->1 ----------
__global__ void classifier_kernel(const float* __restrict__ pooled, const float* __restrict__ cnt,
                                  const float* __restrict__ Wc1, const float* __restrict__ bc1,
                                  const float* __restrict__ Wc2, const float* __restrict__ bc2,
                                  float* __restrict__ out) {
    __shared__ float z[64];
    int gix = blockIdx.x, t = threadIdx.x;             // 64 threads
    float invc = 1.f / fmaxf(cnt[gix], 1.f);
    float acc = bc1[t];
    for (int c = 0; c < 128; ++c) acc += pooled[(size_t)gix * 128 + c] * invc * Wc1[c * 64 + t];
    z[t] = fmaxf(acc, 0.f);
    __syncthreads();
    if (t == 0) {
        float r = bc2[0];
        for (int j = 0; j < 64; ++j) r += z[j] * Wc2[j];
        out[gix] = r;
    }
}

extern "C" void kernel_launch(void* const* d_in, const int* in_sizes, int n_in,
                              void* d_out, int out_size, void* d_ws, size_t ws_size,
                              hipStream_t stream) {
    (void)n_in; (void)out_size; (void)ws_size;
    const int*   x     = (const int*)d_in[0];
    const int*   edge  = (const int*)d_in[1];
    const int*   batch = (const int*)d_in[2];
    const float* emb   = (const float*)d_in[3];
    const int N = in_sizes[0];
    const int E = in_sizes[1] / 2;
    const int* src = edge;
    const int* dst = edge + E;

    // bump allocator over workspace
    char* ws = (char*)d_ws;
    size_t off = 0;
    auto alloc = [&](size_t bytes) -> void* {
        void* p = ws + off;
        off += (bytes + 255) & ~(size_t)255;
        return p;
    };
    unsigned short* hbf  = (unsigned short*)alloc((size_t)N * 256 * 2);
    float* h32   = (float*)alloc((size_t)N * 128 * 4);
    float* q     = (float*)alloc((size_t)N * 1024 * 4);
    float* k     = (float*)alloc((size_t)N * 1024 * 4);
    float* v     = (float*)alloc((size_t)N * 1024 * 4);
    float* xr    = (float*)alloc((size_t)N * 128 * 4);
    float* agg   = (float*)alloc((size_t)N * 1024 * 4);
    float* lgt   = (float*)alloc((size_t)E * 8 * 4);
    unsigned* mkey = (unsigned*)alloc((size_t)N * 8 * 4);
    float* mval  = (float*)alloc((size_t)N * 8 * 4);
    float* den   = (float*)alloc((size_t)N * 8 * 4);
    unsigned short* Wpq = (unsigned short*)alloc((size_t)256 * 1024 * 2);
    unsigned short* Wpk = (unsigned short*)alloc((size_t)256 * 1024 * 2);
    unsigned short* Wpv = (unsigned short*)alloc((size_t)256 * 1024 * 2);
    unsigned short* Wps = (unsigned short*)alloc((size_t)256 * 128 * 2);
    float* pooled = (float*)alloc(64 * 128 * 4);
    float* cnt    = (float*)alloc(64 * 4);

    embed_kernel<<<(int)(((long)N * 256 + 255) / 256), 256, 0, stream>>>(x, emb, hbf, N);

    for (int l = 0; l < 2; ++l) {
        const int K = l ? 128 : 256;
        const int wb = 4 + l * 11;
        const float* Wq = (const float*)d_in[wb + 0];
        const float* bq = (const float*)d_in[wb + 1];
        const float* Wk = (const float*)d_in[wb + 2];
        const float* bk = (const float*)d_in[wb + 3];
        const float* Wv = (const float*)d_in[wb + 4];
        const float* bv = (const float*)d_in[wb + 5];
        const float* Ws = (const float*)d_in[wb + 6];
        const float* bs = (const float*)d_in[wb + 7];
        const float* Wb = (const float*)d_in[wb + 8];
        const float* g  = (const float*)d_in[wb + 9];
        const float* be = (const float*)d_in[wb + 10];

        pack_w_kernel<<<(K * 1024 + 255) / 256, 256, 0, stream>>>(Wq, Wpq, K, 1024);
        pack_w_kernel<<<(K * 1024 + 255) / 256, 256, 0, stream>>>(Wk, Wpk, K, 1024);
        pack_w_kernel<<<(K * 1024 + 255) / 256, 256, 0, stream>>>(Wv, Wpv, K, 1024);
        pack_w_kernel<<<(K * 128  + 255) / 256, 256, 0, stream>>>(Ws, Wps, K, 128);

        const int mtiles = N / 16;                     // N = 50000 -> 3125 exact
        gemm_wmma_kernel<<<dim3(mtiles, 8), 256, 0, stream>>>(hbf, Wpq, bq, q,  N, 1024, K);
        gemm_wmma_kernel<<<dim3(mtiles, 8), 256, 0, stream>>>(hbf, Wpk, bk, k,  N, 1024, K);
        gemm_wmma_kernel<<<dim3(mtiles, 8), 256, 0, stream>>>(hbf, Wpv, bv, v,  N, 1024, K);
        gemm_wmma_kernel<<<dim3(mtiles, 1), 256, 0, stream>>>(hbf, Wps, bs, xr, N, 128,  K);

        hipMemsetAsync(mkey, 0, (size_t)N * 8 * 4, stream);
        hipMemsetAsync(den,  0, (size_t)N * 8 * 4, stream);
        hipMemsetAsync(agg,  0, (size_t)N * 1024 * 4, stream);

        edge_logits_kernel<<<E, 256, 0, stream>>>(q, k, src, dst, lgt, mkey, E);
        decode_m_kernel<<<(N * 8 + 255) / 256, 256, 0, stream>>>(mkey, mval, N * 8);
        ex_den_kernel<<<(E * 8 + 255) / 256, 256, 0, stream>>>(lgt, mval, dst, den, E * 8);
        scatter_kernel<<<E, 256, 0, stream>>>(lgt, den, v, src, dst, agg, E);

        node_finalize_kernel<<<(N + 7) / 8, 256, 0, stream>>>(
            agg, xr, Wb, g, be, l ? h32 : nullptr, h32, hbf, N);
    }

    hipMemsetAsync(pooled, 0, 64 * 128 * 4, stream);
    hipMemsetAsync(cnt,    0, 64 * 4, stream);
    pool_kernel<<<(int)(((long)N * 32 + 255) / 256), 256, 0, stream>>>(h32, batch, pooled, cnt, N);
    classifier_kernel<<<64, 64, 0, stream>>>(
        pooled, cnt,
        (const float*)d_in[26], (const float*)d_in[27],
        (const float*)d_in[28], (const float*)d_in[29],
        (float*)d_out);
}
